// SSM_73031623901714
// MI455X (gfx1250) — matrix operand order repaired and checked by reading, not compiled, not run on hardware
//
#include <hip/hip_runtime.h>
#include <hip/hip_bf16.h>
#include <cstdint>

// ---------------- problem constants ----------------
#define DDIM 1024
#define BBAT 4
#define TSEQ 4096
#define BTD  (BBAT * TSEQ * DDIM)          // 16,777,216
#define ARENA_ROWS 8192                    // 4097 + 2048 + ... + 1
#define NS_ITERS 36
#define EPSV 1e-8f

typedef __attribute__((ext_vector_type(16))) __bf16 v16bf;
typedef __attribute__((ext_vector_type(8)))  float  v8f;
typedef __attribute__((ext_vector_type(2)))  float  v2f;
typedef __attribute__((ext_vector_type(4)))  int    v4i;

// address-space-qualified v4i for the async-to-LDS builtin
typedef __attribute__((address_space(1))) v4i gas_v4i;   // global
typedef __attribute__((address_space(3))) v4i las_v4i;   // LDS

#if defined(__has_builtin)
#if __has_builtin(__builtin_amdgcn_wmma_f32_16x16x4_f32)
#define HAVE_WMMA_F32 1
#endif
#if __has_builtin(__builtin_amdgcn_global_load_async_to_lds_b128)
#define HAVE_ASYNC_LDS 1
#endif
#endif

// dynamic LDS accessor shared by all kernels (single extern symbol)
__device__ inline __hip_bfloat16* dynsmem() {
  extern __shared__ __align__(16) __hip_bfloat16 smem_dyn[];
  return smem_dyn;
}

__device__ inline void async_wait0() {
#if defined(__has_builtin)
#if __has_builtin(__builtin_amdgcn_s_wait_asynccnt)
  __builtin_amdgcn_s_wait_asynccnt(0);
#else
  asm volatile("s_wait_asynccnt 0x0" ::: "memory");
#endif
#endif
}

// ---------------- fragment helpers ----------------
union B16x16 { uint4 u[2]; v16bf v; };

__device__ inline v16bf ld16bf(const __hip_bfloat16* p0, const __hip_bfloat16* p1) {
  B16x16 f;
  f.u[0] = *(const uint4*)p0;
  f.u[1] = *(const uint4*)p1;
  return f.v;
}

__device__ inline v8f wmma_bf16(v16bf a, v16bf b, v8f c) {
  // D = A(16x32) * B(32x16) + C, fp32 accumulate
  return __builtin_amdgcn_wmma_f32_16x16x32_bf16(false, a, false, b, (short)0, c,
                                                 false, false);
}

__device__ inline v8f vzero8() {
  v8f z;
#pragma unroll
  for (int i = 0; i < 8; ++i) z[i] = 0.f;
  return z;
}

// ---------------- small utility kernels ----------------
__global__ void k_zero2(float* p) {
  if (threadIdx.x < 2) p[threadIdx.x] = 0.f;
}

__global__ void k_sqsum(const float* __restrict__ w, float* __restrict__ acc, int n) {
  float s = 0.f;
  for (int i = blockIdx.x * blockDim.x + threadIdx.x; i < n; i += gridDim.x * blockDim.x) {
    float v = w[i];
    s += v * v;
  }
#pragma unroll
  for (int off = 16; off > 0; off >>= 1) s += __shfl_down(s, off, 32);
  if ((threadIdx.x & 31) == 0) atomicAdd(acc, s);
}

__global__ void k_scale(const float* __restrict__ w, float* __restrict__ x,
                        const float* __restrict__ acc, int n) {
  int i = blockIdx.x * blockDim.x + threadIdx.x;
  if (i < n) x[i] = w[i] * rsqrtf(acc[0]);   // ||X||_2 <= ||X||_F = 1
}

__global__ void k_cvt_bf16(const float* __restrict__ src, __hip_bfloat16* __restrict__ dst,
                           int n) {
  int i = blockIdx.x * blockDim.x + threadIdx.x;
  if (i < n) dst[i] = __float2bfloat16(src[i]);
}

// write identity vector into scan level-0 row 0 of each batch arena
__global__ void k_identity(float* __restrict__ arena, const float* __restrict__ idv) {
  int d = blockIdx.x * blockDim.x + threadIdx.x;
  float* A = arena + (size_t)blockIdx.y * ((size_t)ARENA_ROWS * DDIM);
  if (d < DDIM) A[d] = idv[d];
}

// up-sweep odd assignment: R_l[2i+1] = R_{l+1}[i]
__global__ void k_copy_rows(float* __restrict__ arena, int srcBase, int dstBase, int count) {
  size_t idx = (size_t)blockIdx.x * blockDim.x + threadIdx.x;
  size_t tot = (size_t)count * DDIM;
  if (idx >= tot) return;
  int i = (int)(idx / DDIM), d = (int)(idx % DDIM);
  float* A = arena + (size_t)blockIdx.y * ((size_t)ARENA_ROWS * DDIM);
  A[(size_t)(dstBase + 2 * i + 1) * DDIM + d] = A[(size_t)(srcBase + i) * DDIM + d];
}

// ---------------- Newton–Schulz polar iteration (fp32 WMMA) ----------------
// grid (DDIM/16, DDIM/128), block 256 (8 waves). Identical geometry for both paths
// so host launch config is independent of device-side __has_builtin.

// G = X * X^T
__global__ __launch_bounds__(256) void k_ns_gemm_nt(const float* __restrict__ X,
                                                    float* __restrict__ G) {
  const int mBase = blockIdx.x * 16;
  const int cBase = blockIdx.y * 128;
#if defined(HAVE_WMMA_F32)
  const int lane = threadIdx.x & 31, wv = threadIdx.x >> 5;
  const int hf = lane >> 4, l16 = lane & 15;
  const int nTile = cBase + wv * 16;
  const int c0 = hf * 2;                                   // K pair per ISA A/B layout
  const float* ap = X + (size_t)(mBase + l16) * DDIM + c0; // A row
  const float* bp = X + (size_t)(nTile + l16) * DDIM + c0; // B col of X^T = row of X
  v8f acc = vzero8();
  for (int k0 = 0; k0 < DDIM; k0 += 4) {
    v2f a; a[0] = ap[k0]; a[1] = ap[k0 + 1];
    v2f b; b[0] = bp[k0]; b[1] = bp[k0 + 1];
    acc = __builtin_amdgcn_wmma_f32_16x16x4_f32(false, a, false, b, (short)0, acc,
                                                false, false);
  }
#pragma unroll
  for (int g = 0; g < 8; ++g)
    G[(size_t)(mBase + g + hf * 8) * DDIM + nTile + l16] = acc[g];
#else
  for (int e = threadIdx.x; e < 16 * 128; e += 256) {
    int r = mBase + (e >> 7), c = cBase + (e & 127);
    float s = 0.f;
    for (int k = 0; k < DDIM; ++k) s += X[(size_t)r * DDIM + k] * X[(size_t)c * DDIM + k];
    G[(size_t)r * DDIM + c] = s;
  }
#endif
}

// Xout = 1.5*Xin - 0.5*(G*Xin)
__global__ __launch_bounds__(256) void k_ns_update(const float* __restrict__ G,
                                                   const float* __restrict__ Xin,
                                                   float* __restrict__ Xout) {
  const int mBase = blockIdx.x * 16;
  const int cBase = blockIdx.y * 128;
#if defined(HAVE_WMMA_F32)
  const int lane = threadIdx.x & 31, wv = threadIdx.x >> 5;
  const int hf = lane >> 4, l16 = lane & 15;
  const int nTile = cBase + wv * 16;
  const int c0 = hf * 2;
  const float* ap = G + (size_t)(mBase + l16) * DDIM + c0;
  const float* bp = Xin + (size_t)(nTile + l16);           // column access, stride DDIM
  v8f acc = vzero8();
  for (int k0 = 0; k0 < DDIM; k0 += 4) {
    v2f a; a[0] = ap[k0]; a[1] = ap[k0 + 1];
    v2f b; b[0] = bp[(size_t)(k0 + c0) * DDIM];
    b[1] = bp[(size_t)(k0 + c0 + 1) * DDIM];
    acc = __builtin_amdgcn_wmma_f32_16x16x4_f32(false, a, false, b, (short)0, acc,
                                                false, false);
  }
#pragma unroll
  for (int g = 0; g < 8; ++g) {
    size_t o = (size_t)(mBase + g + hf * 8) * DDIM + nTile + l16;
    Xout[o] = 1.5f * Xin[o] - 0.5f * acc[g];
  }
#else
  for (int e = threadIdx.x; e < 16 * 128; e += 256) {
    int r = mBase + (e >> 7), c = cBase + (e & 127);
    float s = 0.f;
    for (int k = 0; k < DDIM; ++k) s += G[(size_t)r * DDIM + k] * Xin[(size_t)k * DDIM + c];
    size_t o = (size_t)r * DDIM + c;
    Xout[o] = 1.5f * Xin[o] - 0.5f * s;
  }
#endif
}

// ---------------- big bf16 GEMM: C(MxN) = A(MxK) * B(NxK)^T ----------------
// block 256 = 8 waves, tile 128x64 (wave: 16 rows x 4 n-tiles).
// B panel (64 x K bf16 = 128KB) staged once in LDS (async-to-LDS when available)
// and shared by all 8 waves -> kills the 8x redundant global B traffic.
// MODE 0: fp32 contiguous dst; MODE 1: bf16 contiguous dst;
// MODE 2: fp32 scattered into scan arena (q rows: batch b, row t+1).
template <int MODE>
__global__ __launch_bounds__(256) void k_gemm_bf16(const __hip_bfloat16* __restrict__ A,
                                                   const __hip_bfloat16* __restrict__ Bw,
                                                   float* __restrict__ dstF,
                                                   __hip_bfloat16* __restrict__ dstB,
                                                   int N, int K) {
  const int tid = threadIdx.x;
  const int lane = tid & 31, wv = tid >> 5;
  const int hf = lane >> 4, l16 = lane & 15;
  const int mBase = blockIdx.x * 128 + wv * 16;
  const int nBase = blockIdx.y * 64;
  __hip_bfloat16* Bs = dynsmem();  // [64][K]

  // ---- stage B panel into LDS ----
  for (int c = tid; c < 64 * (K / 8); c += 256) {
    int r = c / (K / 8);
    int o = (c % (K / 8)) * 8;
    const __hip_bfloat16* gp = Bw + (size_t)(nBase + r) * K + o;
#if defined(HAVE_ASYNC_LDS)
    __builtin_amdgcn_global_load_async_to_lds_b128(
        (gas_v4i*)gp, (las_v4i*)&Bs[r * K + o], 0, 0);
#else
    *(uint4*)&Bs[r * K + o] = *(const uint4*)gp;
#endif
  }
#if defined(HAVE_ASYNC_LDS)
  async_wait0();
#endif
  __syncthreads();

  const __hip_bfloat16* aPtr = A + (size_t)(mBase + l16) * K + hf * 8;
  v8f acc[4];
#pragma unroll
  for (int t = 0; t < 4; ++t) acc[t] = vzero8();

  for (int k0 = 0; k0 < K; k0 += 32) {
    __builtin_prefetch(aPtr + k0 + 128, 0, 1);
    v16bf af = ld16bf(aPtr + k0, aPtr + k0 + 16);
#pragma unroll
    for (int t = 0; t < 4; ++t) {
      const __hip_bfloat16* bPtr = Bs + (size_t)(t * 16 + l16) * K + k0 + hf * 8;
      acc[t] = wmma_bf16(af, ld16bf(bPtr, bPtr + 16), acc[t]);
    }
  }

#pragma unroll
  for (int t = 0; t < 4; ++t) {
#pragma unroll
    for (int g = 0; g < 8; ++g) {
      int m = mBase + g + hf * 8;
      int n = nBase + t * 16 + l16;
      float v = acc[t][g];
      if (MODE == 0) {
        dstF[(size_t)m * N + n] = v;
      } else if (MODE == 1) {
        dstB[(size_t)m * N + n] = __float2bfloat16(v);
      } else {
        int b = m >> 12, tt = m & 4095;  // T = 4096
        dstF[(size_t)b * ((size_t)ARENA_ROWS * DDIM) + (size_t)(tt + 1) * DDIM + n] = v;
      }
    }
  }
}

// ---------------- scan combine: dst = rms(L @ P1^T + R @ P2^T) ----------------
// block 512 = 16 waves, 16 combine-rows x 1024 cols (wave w: cols [64w, 64w+64)).
// L/R rows are converted to bf16 ONCE into LDS and shared by all 16 waves
// (was: 16x redundant global loads + converts). Fused RMS via LDS reduction.
__global__ __launch_bounds__(512) void k_combine_rms(
    float* __restrict__ arena, const __hip_bfloat16* __restrict__ P1b,
    const __hip_bfloat16* __restrict__ P2b, int dstBase, int dstStep, int lBase,
    int lStep, int rBase, int rStep, int count) {
  __shared__ float rowsum[16];
  const int tid = threadIdx.x;
  const int lane = tid & 31, wv = tid >> 5;
  const int hf = lane >> 4, l16 = lane & 15;
  const int i0 = blockIdx.x * 16;
  float* A = arena + (size_t)blockIdx.y * ((size_t)ARENA_ROWS * DDIM);
  __hip_bfloat16* aLs = dynsmem();        // [16][DDIM]
  __hip_bfloat16* aRs = aLs + 16 * DDIM;  // [16][DDIM]

  if (tid < 16) rowsum[tid] = 0.f;

  // ---- stage converted L/R rows into LDS (rows clamped at tail) ----
  for (int e = tid; e < 16 * DDIM; e += 512) {
    int r = e >> 10, d = e & 1023;
    int ri = i0 + r;
    if (ri >= count) ri = count - 1;
    aLs[e] = __float2bfloat16(A[(size_t)(lBase + ri * lStep) * DDIM + d]);
    aRs[e] = __float2bfloat16(A[(size_t)(rBase + ri * rStep) * DDIM + d]);
  }
  __syncthreads();

  const int colBase = wv * 64;
  v8f acc[4];
#pragma unroll
  for (int t = 0; t < 4; ++t) acc[t] = vzero8();

  for (int k0 = 0; k0 < DDIM; k0 += 32) {
    const int s8 = hf * 8;
    const __hip_bfloat16* lp = aLs + l16 * DDIM + k0 + s8;
    const __hip_bfloat16* rp = aRs + l16 * DDIM + k0 + s8;
    v16bf aL = ld16bf(lp, lp + 16);
    v16bf aR = ld16bf(rp, rp + 16);
#pragma unroll
    for (int t = 0; t < 4; ++t) {
      int col = colBase + t * 16 + l16;
      const __hip_bfloat16* b1 = P1b + (size_t)col * DDIM + k0 + hf * 8;
      const __hip_bfloat16* b2 = P2b + (size_t)col * DDIM + k0 + hf * 8;
      acc[t] = wmma_bf16(aL, ld16bf(b1, b1 + 16), acc[t]);
      acc[t] = wmma_bf16(aR, ld16bf(b2, b2 + 16), acc[t]);
    }
  }

  // per-row sum of squares: acc[t][g] belongs to row (g + hf*8), col colBase+t*16+l16
#pragma unroll
  for (int g = 0; g < 8; ++g) {
    float s = 0.f;
#pragma unroll
    for (int t = 0; t < 4; ++t) { float z = acc[t][g]; s += z * z; }
#pragma unroll
    for (int off = 1; off < 16; off <<= 1) s += __shfl_xor(s, off, 32);
    if (l16 == 0) atomicAdd(&rowsum[g + hf * 8], s);
  }
  __syncthreads();

#pragma unroll
  for (int g = 0; g < 8; ++g) {
    int rloc = g + hf * 8;
    int gi = i0 + rloc;
    if (gi < count) {
      float sc = rsqrtf(rowsum[rloc] * (1.0f / DDIM) + EPSV);
      float* drow = A + (size_t)(dstBase + gi * dstStep) * DDIM;
#pragma unroll
      for (int t = 0; t < 4; ++t) drow[colBase + t * 16 + l16] = acc[t][g] * sc;
    }
  }
}

// ---------------- finalize: qs -> d_out, u = bf16(qs * v) ----------------
__global__ void k_finish_qs(const float* __restrict__ arena,
                            const __hip_bfloat16* __restrict__ vb,
                            float* __restrict__ outQs, __hip_bfloat16* __restrict__ ub) {
  size_t idx = (size_t)blockIdx.x * blockDim.x + threadIdx.x;
  if (idx >= (size_t)BTD) return;
  int b = (int)(idx >> 22);               // T*D = 2^22
  int t = (int)((idx >> 10) & 4095);
  int d = (int)(idx & 1023);
  float q = arena[(size_t)b * ((size_t)ARENA_ROWS * DDIM) + (size_t)(t + 1) * DDIM + d];
  float v = __bfloat162float(vb[idx]);
  outQs[idx] = q;
  ub[idx] = __float2bfloat16(q * v);
}

// ---------------- host orchestration ----------------
extern "C" void kernel_launch(void* const* d_in, const int* in_sizes, int n_in,
                              void* d_out, int out_size, void* d_ws, size_t ws_size,
                              hipStream_t stream) {
  const float* x   = (const float*)d_in[0];
  const float* Wq  = (const float*)d_in[1];
  const float* Wv  = (const float*)d_in[2];
  const float* Wc  = (const float*)d_in[3];
  const float* Wp1 = (const float*)d_in[4];
  const float* Wp2 = (const float*)d_in[5];
  const float* idv = (const float*)d_in[6];
  float* out = (float*)d_out;  // [Y (BTD) | qs (BTD)]

  // workspace carve-out
  size_t off = 0;
  char* base = (char*)d_ws;
  auto take = [&](size_t bytes) -> void* {
    off = (off + 255) & ~(size_t)255;
    void* p = base + off;
    off += bytes;
    return p;
  };
  const size_t DD = (size_t)DDIM * DDIM;
  float* norm = (float*)take(2 * sizeof(float));
  float* G  = (float*)take(DD * 4);
  float* Xa = (float*)take(DD * 4);
  float* Xb = (float*)take(DD * 4);
  __hip_bfloat16* p1b = (__hip_bfloat16*)take(DD * 2);
  __hip_bfloat16* p2b = (__hip_bfloat16*)take(DD * 2);
  __hip_bfloat16* wqb = (__hip_bfloat16*)take(DD * 2);
  __hip_bfloat16* wvb = (__hip_bfloat16*)take(DD * 2);
  __hip_bfloat16* wcb = (__hip_bfloat16*)take(DD * 2);
  __hip_bfloat16* xb  = (__hip_bfloat16*)take((size_t)BTD * 2);
  __hip_bfloat16* vb  = (__hip_bfloat16*)take((size_t)BTD * 2);
  float* arena = (float*)take((size_t)BBAT * ARENA_ROWS * DDIM * 4);
  __hip_bfloat16* ub = xb;  // x no longer needed once q,v exist

  const dim3 blk256(256), blk512(512);
  const dim3 nsGrid(DDIM / 16, DDIM / 128);             // (64, 8)
  const dim3 gemmGrid((BBAT * TSEQ) / 128, DDIM / 64);  // (128, 16)
  const size_t gemmLds = 64 * (size_t)DDIM * 2;         // 128 KB B panel
  const size_t combLds = 2 * 16 * (size_t)DDIM * 2;     // 64 KB staged L/R rows

  // --- polar factors via Newton–Schulz (fp32 WMMA) ---
  k_zero2<<<1, 32, 0, stream>>>(norm);
  k_sqsum<<<512, blk256, 0, stream>>>(Wp1, norm + 0, (int)DD);
  k_sqsum<<<512, blk256, 0, stream>>>(Wp2, norm + 1, (int)DD);

  for (int mat = 0; mat < 2; ++mat) {
    const float* W = mat ? Wp2 : Wp1;
    __hip_bfloat16* pb = mat ? p2b : p1b;
    k_scale<<<(int)(DD / 256), blk256, 0, stream>>>(W, Xa, norm + mat, (int)DD);
    float* Xc = Xa; float* Xn = Xb;
    for (int it = 0; it < NS_ITERS; ++it) {
      k_ns_gemm_nt<<<nsGrid, blk256, 0, stream>>>(Xc, G);
      k_ns_update<<<nsGrid, blk256, 0, stream>>>(G, Xc, Xn);
      float* tmp = Xc; Xc = Xn; Xn = tmp;
    }
    k_cvt_bf16<<<(int)(DD / 256), blk256, 0, stream>>>(Xc, pb, (int)DD);
  }

  // --- bf16 operand copies ---
  k_cvt_bf16<<<(int)(DD / 256), blk256, 0, stream>>>(Wq, wqb, (int)DD);
  k_cvt_bf16<<<(int)(DD / 256), blk256, 0, stream>>>(Wv, wvb, (int)DD);
  k_cvt_bf16<<<(int)(DD / 256), blk256, 0, stream>>>(Wc, wcb, (int)DD);
  k_cvt_bf16<<<BTD / 256, blk256, 0, stream>>>(x, xb, BTD);

  // --- q -> arena level-0 rows 1..T ; v -> bf16 ---
  k_gemm_bf16<2><<<gemmGrid, blk256, gemmLds, stream>>>(xb, wqb, arena, nullptr, DDIM, DDIM);
  k_gemm_bf16<1><<<gemmGrid, blk256, gemmLds, stream>>>(xb, wvb, nullptr, vb, DDIM, DDIM);
  k_identity<<<dim3(DDIM / 256, BBAT), blk256, 0, stream>>>(arena, idv);

  // --- JAX-order associative scan (combine is non-associative -> order matters) ---
  static const int nlv[13] = {4097, 2048, 1024, 512, 256, 128, 64, 32, 16, 8, 4, 2, 1};
  int lvlOff[13];
  lvlOff[0] = 0;
  for (int l = 0; l < 12; ++l) lvlOff[l + 1] = lvlOff[l] + nlv[l];

  // down-sweep: S_{l+1}[i] = combine(S_l[2i], S_l[2i+1])
  for (int l = 0; l < 12; ++l) {
    int cnt = nlv[l + 1];
    k_combine_rms<<<dim3((cnt + 15) / 16, BBAT), blk512, combLds, stream>>>(
        arena, p1b, p2b, lvlOff[l + 1], 1, lvlOff[l], 2, lvlOff[l] + 1, 2, cnt);
  }
  // up-sweep: R_l[2i+1] = R_{l+1}[i]; R_l[2i] = combine(R_{l+1}[i-1], S_l[2i]), i>=1
  for (int l = 11; l >= 0; --l) {
    int ccopy = nlv[l + 1];
    k_copy_rows<<<dim3((int)(((size_t)ccopy * DDIM + 255) / 256), BBAT), blk256, 0,
                  stream>>>(arena, lvlOff[l + 1], lvlOff[l], ccopy);
    int cup = (nlv[l] - 1) / 2;
    if (cup > 0) {
      k_combine_rms<<<dim3((cup + 15) / 16, BBAT), blk512, combLds, stream>>>(
          arena, p1b, p2b, lvlOff[l] + 2, 2, lvlOff[l + 1], 1, lvlOff[l] + 2, 2, cup);
    }
  }

  // --- qs out + u = qs*v ; Y = u @ Wc^T ---
  k_finish_qs<<<BTD / 256, blk256, 0, stream>>>(arena, vb, out + (size_t)BTD, ub);
  k_gemm_bf16<0><<<gemmGrid, blk256, gemmLds, stream>>>(ub, wcb, out, nullptr, DDIM, DDIM);
}
